// PCA_GM_Net_18829136625932
// MI455X (gfx1250) — compile-verified, matrix operand order checked by reading, not run on hardware
//
#include <hip/hip_runtime.h>

// ---------------------------------------------------------------------------
// PCA-GM forward for MI455X (gfx1250).
// All GEMMs via v_wmma_f32_16x16x32_bf16 (f32 accumulate), double-buffered LDS
// staging with aligned b128 copies only (operands pre-arranged so A is M x K
// and B is N x K row-major everywhere). Sinkhorn/normalize glue in f32 VALU.
// ---------------------------------------------------------------------------

typedef __attribute__((ext_vector_type(16))) __bf16 v16bf;
typedef __attribute__((ext_vector_type(8)))  __bf16 v8bf;
typedef __attribute__((ext_vector_type(8)))  float  v8f;

__device__ __forceinline__ unsigned short f2bf(float x) {
  unsigned int u = __float_as_uint(x);
  unsigned int r = u + 0x7FFFu + ((u >> 16) & 1u);   // round-to-nearest-even
  return (unsigned short)(r >> 16);
}
__device__ __forceinline__ float bf2f(unsigned short h) {
  return __uint_as_float(((unsigned int)h) << 16);
}

// ---------------------------------------------------------------------------
// WMMA GEMM: out = act(A @ B^T + bias + C0)
//   A: bf16, M x K row-major [per-batch stride sA]
//   B: bf16, N x K row-major [per-batch stride sB]   (i.e. B^T of the math op)
//   C0: optional bf16 M x N addend (natural layout); bias: optional f32 [N]
//   Outputs (any subset): outF f32 MxN (ldOF), outH bf16 MxN (ldOH),
//                         outHT bf16 transposed N x M (ldOHT).
// Requires M%128==0, N%128==0, K%32==0, all lds multiples of 8.
// ---------------------------------------------------------------------------
#define BM 128
#define BN 128
#define BK 32
#define LK 48   // padded LDS K-stride (halfs); 96B rows keep 16B-aligned frags

__global__ __launch_bounds__(256)
void gemm_wmma_bf16(const unsigned short* __restrict__ Ag, long long sA, int ldA,
                    const unsigned short* __restrict__ Bg, long long sB, int ldB,
                    const float* __restrict__ bias,
                    const unsigned short* __restrict__ C0, long long sC0, int ldC0,
                    float* __restrict__ outF, long long sOF, int ldOF,
                    unsigned short* __restrict__ outH, long long sOH, int ldOH,
                    unsigned short* __restrict__ outHT, long long sOHT, int ldOHT,
                    int K, int doRelu)
{
  __shared__ unsigned short As[2][BM * LK];
  __shared__ unsigned short Bs[2][BN * LK];

  const int z = blockIdx.z;
  Ag += (long long)z * sA;
  Bg += (long long)z * sB;
  if (C0) C0 += (long long)z * sC0;

  const int blkM = blockIdx.y * BM;
  const int blkN = blockIdx.x * BN;

  const int t    = threadIdx.x;
  const int lane = t & 31;
  const int wave = t >> 5;
  const int wm0  = (wave >> 1) * 32;   // 4 waves along M
  const int wn0  = (wave & 1) * 64;    // 2 waves along N
  const int hl   = lane >> 4;          // half-wave select
  const int r16  = lane & 15;

  const int rowS = t >> 1;             // staging row 0..127
  const int colS = (t & 1) << 4;       // staging col 0 or 16

  const unsigned short* aSrc = Ag + (long long)(blkM + rowS) * ldA + colS;
  const unsigned short* bSrc = Bg + (long long)(blkN + rowS) * ldB + colS;

  v8f acc[2][4];
#pragma unroll
  for (int i = 0; i < 2; ++i)
#pragma unroll
    for (int j = 0; j < 4; ++j) {
      v8f zz = {0.f, 0.f, 0.f, 0.f, 0.f, 0.f, 0.f, 0.f};
      acc[i][j] = zz;
    }

  auto stage = [&](int k0, int buf) {
    const uint4* sa = reinterpret_cast<const uint4*>(aSrc + k0);
    uint4 a0 = sa[0], a1 = sa[1];
    const uint4* sb = reinterpret_cast<const uint4*>(bSrc + k0);
    uint4 b0 = sb[0], b1 = sb[1];
    uint4* da = reinterpret_cast<uint4*>(&As[buf][rowS * LK + colS]);
    da[0] = a0; da[1] = a1;
    uint4* db = reinterpret_cast<uint4*>(&Bs[buf][rowS * LK + colS]);
    db[0] = b0; db[1] = b1;
  };

  stage(0, 0);
  __syncthreads();

  int buf = 0;
  for (int k0 = 0; k0 < K; k0 += BK) {
    if (k0 + BK < K) stage(k0 + BK, buf ^ 1);           // fill other buffer
    if (k0 + 2 * BK < K) {                              // gfx1250 global_prefetch
      __builtin_prefetch(aSrc + k0 + 2 * BK, 0, 3);
      __builtin_prefetch(bSrc + k0 + 2 * BK, 0, 3);
    }

    // fragments per ISA 16-bit layout: lanes<16 hold K 0-7 & 16-23,
    // lanes>=16 hold K 8-15 & 24-31, for row (lane&15)
    v16bf afr[2], bfr[4];
#pragma unroll
    for (int i = 0; i < 2; ++i) {
      const unsigned short* base = &As[buf][(wm0 + i * 16 + r16) * LK + hl * 8];
      v8bf lo = *reinterpret_cast<const v8bf*>(base);
      v8bf hi = *reinterpret_cast<const v8bf*>(base + 16);
#pragma unroll
      for (int e = 0; e < 8; ++e) { afr[i][e] = lo[e]; afr[i][e + 8] = hi[e]; }
    }
#pragma unroll
    for (int j = 0; j < 4; ++j) {
      const unsigned short* base = &Bs[buf][(wn0 + j * 16 + r16) * LK + hl * 8];
      v8bf lo = *reinterpret_cast<const v8bf*>(base);
      v8bf hi = *reinterpret_cast<const v8bf*>(base + 16);
#pragma unroll
      for (int e = 0; e < 8; ++e) { bfr[j][e] = lo[e]; bfr[j][e + 8] = hi[e]; }
    }
#pragma unroll
    for (int i = 0; i < 2; ++i)
#pragma unroll
      for (int j = 0; j < 4; ++j)
        acc[i][j] = __builtin_amdgcn_wmma_f32_16x16x32_bf16(
            false, afr[i], false, bfr[j], (short)0, acc[i][j], false, false);

    __syncthreads();
    buf ^= 1;
  }

  // ---- epilogue: C/D layout VGPR v -> M = v + 8*hl, N = lane&15 ----
#pragma unroll
  for (int i = 0; i < 2; ++i) {
#pragma unroll
    for (int j = 0; j < 4; ++j) {
      const int colG = blkN + wn0 + j * 16 + r16;
      const float bv = bias ? bias[colG] : 0.0f;
#pragma unroll
      for (int v = 0; v < 8; ++v) {
        const int rowG = blkM + wm0 + i * 16 + hl * 8 + v;
        float val = acc[i][j][v] + bv;
        if (C0) val += bf2f(C0[(long long)rowG * ldC0 + colG]);
        if (doRelu) val = fmaxf(val, 0.0f);
        if (outF)  outF[(long long)z * sOF + (long long)rowG * ldOF + colG] = val;
        if (outH)  outH[(long long)z * sOH + (long long)rowG * ldOH + colG] = f2bf(val);
        if (outHT) outHT[(long long)z * sOHT + (long long)colG * ldOHT + rowG] = f2bf(val);
      }
    }
  }
}

// ---------------------------------------------------------------------------
// Elementwise f32 -> bf16 (natural layout)
// ---------------------------------------------------------------------------
__global__ void cvt_f32_bf16(const float* __restrict__ x,
                             unsigned short* __restrict__ y, long long n) {
  long long i = (long long)blockIdx.x * blockDim.x + threadIdx.x;
  const long long st = (long long)gridDim.x * blockDim.x;
  for (; i < n; i += st) y[i] = f2bf(x[i]);
}

// ---------------------------------------------------------------------------
// Transposing f32 (R x C) -> bf16 (C x R), 32x32 LDS tiles. R,C % 32 == 0.
// grid = (C/32, R/32), block = 256 (8 rows of 32).
// ---------------------------------------------------------------------------
__global__ __launch_bounds__(256)
void cvt_t_f32_bf16(const float* __restrict__ x, unsigned short* __restrict__ y,
                    int R, int C) {
  __shared__ float tile[32][33];
  const int c0 = blockIdx.x * 32, r0 = blockIdx.y * 32;
  const int tx = threadIdx.x & 31, ty = threadIdx.x >> 5;
#pragma unroll
  for (int i = 0; i < 32; i += 8)
    tile[ty + i][tx] = x[(long long)(r0 + ty + i) * C + c0 + tx];
  __syncthreads();
#pragma unroll
  for (int i = 0; i < 32; i += 8)
    y[(long long)(c0 + ty + i) * R + r0 + tx] = f2bf(tile[tx][ty + i]);
}

// ---------------------------------------------------------------------------
// Column-wise L1 normalize of A (B,256,256) -> bf16.  One block per batch.
// ---------------------------------------------------------------------------
__global__ __launch_bounds__(256)
void anorm_kernel(const float* __restrict__ A, unsigned short* __restrict__ out) {
  const int b = blockIdx.x;
  const int m = threadIdx.x;
  const float* Ab = A + (long long)b * 65536;
  float s = 0.f;
  for (int n = 0; n < 256; ++n) s += fabsf(Ab[n * 256 + m]);
  const float inv = 1.0f / fmaxf(s, 1e-12f);
  unsigned short* ob = out + (long long)b * 65536;
  for (int n = 0; n < 256; ++n) ob[n * 256 + m] = f2bf(Ab[n * 256 + m] * inv);
}

// ---------------------------------------------------------------------------
// Sinkhorn pieces (B=32, N=M=256), mask from device n1/n2, NEG=-1e9, tau=0.05
// ---------------------------------------------------------------------------
__global__ void sk_init(float* __restrict__ s, const int* __restrict__ n1,
                        const int* __restrict__ n2) {
  const int idx = blockIdx.x * 256 + threadIdx.x;        // grid = 8192 blocks
  const int b = idx >> 16, n = (idx >> 8) & 255, m = idx & 255;
  const bool ok = (n < n1[b]) && (m < n2[b]);
  s[idx] = ok ? s[idx] * 20.0f : -1e9f;
}

__global__ __launch_bounds__(256)
void sk_norm(float* __restrict__ s, const int* __restrict__ n1,
             const int* __restrict__ n2, int axis) {
  const int rowId = blockIdx.x * 8 + (threadIdx.x >> 5); // grid = 1024 blocks
  const int lane  = threadIdx.x & 31;
  const int b = rowId >> 8, r = rowId & 255;
  long long base; int stride;
  if (axis == 2) { base = (long long)b * 65536 + (long long)r * 256; stride = 1; }
  else           { base = (long long)b * 65536 + r;                  stride = 256; }
  float v[8];
#pragma unroll
  for (int e = 0; e < 8; ++e) v[e] = s[base + (long long)(lane + e * 32) * stride];
  float mx = v[0];
#pragma unroll
  for (int e = 1; e < 8; ++e) mx = fmaxf(mx, v[e]);
  for (int o = 16; o; o >>= 1) mx = fmaxf(mx, __shfl_xor(mx, o, 32));
  float sum = 0.f;
#pragma unroll
  for (int e = 0; e < 8; ++e) sum += expf(v[e] - mx);
  for (int o = 16; o; o >>= 1) sum += __shfl_xor(sum, o, 32);
  const float lse = mx + logf(sum);
  const int nn1 = n1[b], nn2 = n2[b];
#pragma unroll
  for (int e = 0; e < 8; ++e) {
    const int o = lane + e * 32;
    const int nI = (axis == 2) ? r : o;
    const int mI = (axis == 2) ? o : r;
    const bool ok = (nI < nn1) && (mI < nn2);
    s[base + (long long)o * stride] = ok ? (v[e] - lse) : -1e9f;
  }
}

__global__ void sk_exp(const float* __restrict__ s, float* __restrict__ oF,
                       unsigned short* __restrict__ oH,
                       unsigned short* __restrict__ oHT) {
  const int idx = blockIdx.x * 256 + threadIdx.x;        // grid = 8192 blocks
  const float v = expf(s[idx]);
  if (oF) oF[idx] = v;
  if (oH) oH[idx] = f2bf(v);
  if (oHT) {
    const int b = idx >> 16, n = (idx >> 8) & 255, m = idx & 255;
    oHT[(b << 16) + (m << 8) + n] = f2bf(v);
  }
}

// ---------------------------------------------------------------------------
// Host driver
// ---------------------------------------------------------------------------
extern "C" void kernel_launch(void* const* d_in, const int* in_sizes, int n_in,
                              void* d_out, int out_size, void* d_ws, size_t ws_size,
                              hipStream_t stream) {
  const float* feat1 = (const float*)d_in[0];
  const float* feat2 = (const float*)d_in[1];
  const float* A1    = (const float*)d_in[2];
  const float* A2    = (const float*)d_in[3];
  const int*   n1    = (const int*)d_in[4];
  const int*   n2    = (const int*)d_in[5];
  // d_in[6] cross_iter_num (<=0), d_in[7] sk_max_iter (==10, fixed for capture)
  const float* W0a = (const float*)d_in[8];
  const float* b0a = (const float*)d_in[9];
  const float* W0u = (const float*)d_in[10];
  const float* b0u = (const float*)d_in[11];
  const float* W1a = (const float*)d_in[12];
  const float* b1a = (const float*)d_in[13];
  const float* W1u = (const float*)d_in[14];
  const float* b1u = (const float*)d_in[15];
  const float* Wc  = (const float*)d_in[16];
  const float* bc  = (const float*)d_in[17];
  const float* Af0 = (const float*)d_in[18];
  const float* Af1 = (const float*)d_in[19];

  const long long E_W0 = 1024LL * 2048;          // 2M
  const long long E_W1 = 2048LL * 2048;          // 4M
  const long long E_Wc = 4096LL * 2048;          // 8M
  const long long E_F  = 32LL * 256 * 1024;      // 8M
  const long long E_A  = 32LL * 256 * 256;       // 2M
  const long long E_E  = 32LL * 256 * 2048;      // 16M

  char* ws = (char*)d_ws;
  size_t off = 0;
  auto alloc = [&](size_t bytes) {
    size_t r = off; off = (off + bytes + 255) & ~(size_t)255; return r;
  };
  // ~432 MB total scratch. All weight copies stored TRANSPOSED (N x K).
  unsigned short* pW0a = (unsigned short*)(ws + alloc(E_W0 * 2));  // 2048x1024
  unsigned short* pW0u = (unsigned short*)(ws + alloc(E_W0 * 2));
  unsigned short* pW1a = (unsigned short*)(ws + alloc(E_W1 * 2));  // 2048x2048
  unsigned short* pW1u = (unsigned short*)(ws + alloc(E_W1 * 2));
  unsigned short* pWc  = (unsigned short*)(ws + alloc(E_Wc * 2));  // 2048x4096
  unsigned short* pAf0 = (unsigned short*)(ws + alloc(E_W1 * 2));
  unsigned short* pAf1 = (unsigned short*)(ws + alloc(E_W1 * 2));
  unsigned short* pF1  = (unsigned short*)(ws + alloc(E_F * 2));   // natural
  unsigned short* pF2  = (unsigned short*)(ws + alloc(E_F * 2));
  unsigned short* pAn1 = (unsigned short*)(ws + alloc(E_A * 2));
  unsigned short* pAn2 = (unsigned short*)(ws + alloc(E_A * 2));
  unsigned short* pAx  = (unsigned short*)(ws + alloc(E_E * 2));   // TRANSPOSED 2048x8192
  unsigned short* pUx  = (unsigned short*)(ws + alloc(E_E * 2));   // natural
  unsigned short* pE1  = (unsigned short*)(ws + alloc(E_E * 2));   // natural
  unsigned short* pE2  = (unsigned short*)(ws + alloc(E_E * 2));
  unsigned short* pE1T = (unsigned short*)(ws + alloc(E_E * 2));   // per-batch 2048x256
  unsigned short* pE2T = (unsigned short*)(ws + alloc(E_E * 2));
  unsigned short* pTmp = (unsigned short*)(ws + alloc(E_E * 2));
  float*          pS   = (float*)(ws + alloc(E_A * 4));
  unsigned short* pSbf = (unsigned short*)(ws + alloc(E_A * 2));
  unsigned short* pSbT = (unsigned short*)(ws + alloc(E_A * 2));
  unsigned short* pX1  = (unsigned short*)(ws + alloc(E_E * 2));
  unsigned short* pX2  = (unsigned short*)(ws + alloc(E_E * 2));
  unsigned short* pP   = (unsigned short*)(ws + alloc(E_E * 2));

  auto cvt = [&](const float* x, unsigned short* y, long long n) {
    cvt_f32_bf16<<<4096, 256, 0, stream>>>(x, y, n);
  };
  auto cvtT = [&](const float* x, unsigned short* y, int R, int C) {
    dim3 grid(C / 32, R / 32);
    cvt_t_f32_bf16<<<grid, 256, 0, stream>>>(x, y, R, C);
  };
  // out = act(A @ B^T + bias + C0); A: MxK (ldA), B: NxK (ldB)
  auto G = [&](const unsigned short* A, long long sA, int ldA,
               const unsigned short* B, long long sB, int ldB,
               const float* bias, const unsigned short* C0, long long sC0, int ldC0,
               float* oF, long long sOF, int ldOF,
               unsigned short* oH, long long sOH, int ldOH,
               unsigned short* oHT, long long sOHT, int ldOHT,
               int M, int N, int K, int relu, int batch) {
    dim3 grid(N / 128, M / 128, batch), block(256);
    gemm_wmma_bf16<<<grid, block, 0, stream>>>(A, sA, ldA, B, sB, ldB, bias,
        C0, sC0, ldC0, oF, sOF, ldOF, oH, sOH, ldOH, oHT, sOHT, ldOHT, K, relu);
  };
  auto sinkhorn = [&](float* s, float* oF, unsigned short* oH, unsigned short* oHT) {
    sk_init<<<8192, 256, 0, stream>>>(s, n1, n2);
    for (int i = 0; i < 10; ++i)
      sk_norm<<<1024, 256, 0, stream>>>(s, n1, n2, (i % 2 == 0) ? 2 : 1);
    sk_exp<<<8192, 256, 0, stream>>>(s, oF, oH, oHT);
  };

  // ---- phase 0: bf16 conversions (weights transposed) + adjacency norm ----
  cvtT(W0a, pW0a, 1024, 2048); cvtT(W0u, pW0u, 1024, 2048);
  cvtT(W1a, pW1a, 2048, 2048); cvtT(W1u, pW1u, 2048, 2048);
  cvtT(Wc,  pWc,  4096, 2048);
  cvtT(Af0, pAf0, 2048, 2048); cvtT(Af1, pAf1, 2048, 2048);
  cvt(feat1, pF1, E_F); cvt(feat2, pF2, E_F);
  anorm_kernel<<<32, 256, 0, stream>>>(A1, pAn1);
  anorm_kernel<<<32, 256, 0, stream>>>(A2, pAn2);

  const long long sE = 256LL * 2048;   // per-batch stride of (256,2048) tiles
  const long long sS = 65536;          // per-batch stride of (256,256) tiles
  unsigned short* pWcT = pWc;                         // rows 0..2047 of Wc (K-offset 0)
  unsigned short* pWcB = pWc + 2048;                  // rows 2048..4095 (K-offset 2048)

  // ---- layer 0 gconv (Siamese) ----
  // graph1: ax^T -> pAx (2048x8192), ux -> pUx, emb1 = An1@ax + ux
  G(pF1, 0, 1024, pW0a, 0, 1024, b0a, nullptr, 0, 0,
    nullptr, 0, 0, nullptr, 0, 0, pAx, 0, 8192, 8192, 2048, 1024, 1, 1);
  G(pF1, 0, 1024, pW0u, 0, 1024, b0u, nullptr, 0, 0,
    nullptr, 0, 0, pUx, 0, 2048, nullptr, 0, 0, 8192, 2048, 1024, 1, 1);
  G(pAn1, sS, 256, pAx, 256, 8192, nullptr, pUx, sE, 2048,
    nullptr, 0, 0, pE1, sE, 2048, pE1T, sE, 256, 256, 2048, 256, 0, 32);
  // graph2
  G(pF2, 0, 1024, pW0a, 0, 1024, b0a, nullptr, 0, 0,
    nullptr, 0, 0, nullptr, 0, 0, pAx, 0, 8192, 8192, 2048, 1024, 1, 1);
  G(pF2, 0, 1024, pW0u, 0, 1024, b0u, nullptr, 0, 0,
    nullptr, 0, 0, pUx, 0, 2048, nullptr, 0, 0, 8192, 2048, 1024, 1, 1);
  G(pAn2, sS, 256, pAx, 256, 8192, nullptr, pUx, sE, 2048,
    nullptr, 0, 0, pE2, sE, 2048, pE2T, sE, 256, 256, 2048, 256, 0, 32);

  // ---- affinity 0: s = (emb1 @ Aaff0) @ emb2^T ; sinkhorn -> bf16 + bf16^T ----
  G(pE1, 0, 2048, pAf0, 0, 2048, nullptr, nullptr, 0, 0,
    nullptr, 0, 0, pTmp, 0, 2048, nullptr, 0, 0, 8192, 2048, 2048, 0, 1);
  G(pTmp, sE, 2048, pE2, sE, 2048, nullptr, nullptr, 0, 0,
    pS, sS, 256, nullptr, 0, 0, nullptr, 0, 0, 256, 256, 2048, 0, 32);
  sinkhorn(pS, nullptr, pSbf, pSbT);

  // ---- cross-graph update: concat-GEMM via split-K over Wc ----
  G(pSbf, sS, 256, pE2T, sE, 256, nullptr, nullptr, 0, 0,
    nullptr, 0, 0, pX1, sE, 2048, nullptr, 0, 0, 256, 2048, 256, 0, 32); // s @ emb2
  G(pSbT, sS, 256, pE1T, sE, 256, nullptr, nullptr, 0, 0,
    nullptr, 0, 0, pX2, sE, 2048, nullptr, 0, 0, 256, 2048, 256, 0, 32); // s^T @ emb1
  // new1 = emb1@Wc_top + bc, then += cross1@Wc_bot
  G(pE1, 0, 2048, pWcT, 0, 4096, bc, nullptr, 0, 0,
    nullptr, 0, 0, pP, 0, 2048, nullptr, 0, 0, 8192, 2048, 2048, 0, 1);
  G(pX1, 0, 2048, pWcB, 0, 4096, nullptr, pP, 0, 2048,
    nullptr, 0, 0, pE1, 0, 2048, nullptr, 0, 0, 8192, 2048, 2048, 0, 1);
  // new2 = emb2@Wc_top + bc, then += cross2@Wc_bot
  G(pE2, 0, 2048, pWcT, 0, 4096, bc, nullptr, 0, 0,
    nullptr, 0, 0, pP, 0, 2048, nullptr, 0, 0, 8192, 2048, 2048, 0, 1);
  G(pX2, 0, 2048, pWcB, 0, 4096, nullptr, pP, 0, 2048,
    nullptr, 0, 0, pE2, 0, 2048, nullptr, 0, 0, 8192, 2048, 2048, 0, 1);

  // ---- layer 1 gconv ----
  G(pE1, 0, 2048, pW1a, 0, 2048, b1a, nullptr, 0, 0,
    nullptr, 0, 0, nullptr, 0, 0, pAx, 0, 8192, 8192, 2048, 2048, 1, 1);
  G(pE1, 0, 2048, pW1u, 0, 2048, b1u, nullptr, 0, 0,
    nullptr, 0, 0, pUx, 0, 2048, nullptr, 0, 0, 8192, 2048, 2048, 1, 1);
  G(pAn1, sS, 256, pAx, 256, 8192, nullptr, pUx, sE, 2048,
    nullptr, 0, 0, pX1, sE, 2048, nullptr, 0, 0, 256, 2048, 256, 0, 32); // emb1'
  G(pE2, 0, 2048, pW1a, 0, 2048, b1a, nullptr, 0, 0,
    nullptr, 0, 0, nullptr, 0, 0, pAx, 0, 8192, 8192, 2048, 2048, 1, 1);
  G(pE2, 0, 2048, pW1u, 0, 2048, b1u, nullptr, 0, 0,
    nullptr, 0, 0, pUx, 0, 2048, nullptr, 0, 0, 8192, 2048, 2048, 1, 1);
  G(pAn2, sS, 256, pAx, 256, 8192, nullptr, pUx, sE, 2048,
    nullptr, 0, 0, pX2, sE, 2048, nullptr, 0, 0, 256, 2048, 256, 0, 32); // emb2'

  // ---- affinity 1 + final sinkhorn -> d_out (f32, 32x256x256) ----
  G(pX1, 0, 2048, pAf1, 0, 2048, nullptr, nullptr, 0, 0,
    nullptr, 0, 0, pTmp, 0, 2048, nullptr, 0, 0, 8192, 2048, 2048, 0, 1);
  G(pTmp, sE, 2048, pX2, sE, 2048, nullptr, nullptr, 0, 0,
    pS, sS, 256, nullptr, 0, 0, nullptr, 0, 0, 256, 256, 2048, 0, 32);
  sinkhorn(pS, (float*)d_out, nullptr, nullptr);
}
